// Net_51539607823
// MI455X (gfx1250) — matrix operand order repaired and checked by reading, not compiled, not run on hardware
//
#include <hip/hip_runtime.h>
#include <math.h>

#define N0 1000000
#define N1 200000
#define N2 25000
#define E1 3200000
#define E2 400000
// D_IN = 128, D_HID = 16, D_OUT = 14

typedef __attribute__((ext_vector_type(16))) _Float16 v16h;
typedef __attribute__((ext_vector_type(8)))  float    v8f;

// Load 16 contiguous f32 and convert to a packed v16h (one B-matrix K-row).
static __device__ __forceinline__ v16h load_row16_f16(const float* __restrict__ p) {
    float4 f0 = *(const float4*)(p);
    float4 f1 = *(const float4*)(p + 4);
    float4 f2 = *(const float4*)(p + 8);
    float4 f3 = *(const float4*)(p + 12);
    v16h b;
    b[0]=(_Float16)f0.x;  b[1]=(_Float16)f0.y;  b[2]=(_Float16)f0.z;  b[3]=(_Float16)f0.w;
    b[4]=(_Float16)f1.x;  b[5]=(_Float16)f1.y;  b[6]=(_Float16)f1.z;  b[7]=(_Float16)f1.w;
    b[8]=(_Float16)f2.x;  b[9]=(_Float16)f2.y;  b[10]=(_Float16)f2.z; b[11]=(_Float16)f2.w;
    b[12]=(_Float16)f3.x; b[13]=(_Float16)f3.y; b[14]=(_Float16)f3.z; b[15]=(_Float16)f3.w;
    return b;
}

// ---------------------------------------------------------------------------
// Kernel 1: y = x @ W1l for all N0 rows; xr = x @ W1r for rows < N1.
// One wave32 per 16-row tile. K=128 -> 4 WMMA steps of K=32.
// A layout (16-bit A 16x32): lane L<16 holds M=L, K {kb..kb+7, kb+16..kb+23};
// lanes 16..31 hold M=L-16, K {kb+8..kb+15, kb+24..kb+31}.
// B layout: lane&15 = column N, v16h element s = K row in the same k order as A.
// Tile index is forced into an SGPR (readfirstlane) so the rows>=N0 early-out
// and the doR guard compile to scalar branches: EXEC is all-1s at every WMMA
// that issues, as the ISA requires.
// ---------------------------------------------------------------------------
__global__ __launch_bounds__(256) void k_gemm1(const float* __restrict__ x,
                                               const float* __restrict__ Wl,
                                               const float* __restrict__ Wr,
                                               float* __restrict__ y,
                                               float* __restrict__ xr) {
    const int  lane = threadIdx.x & 31;
    const int  wid  = __builtin_amdgcn_readfirstlane(
                          blockIdx.x * (blockDim.x >> 5) + (threadIdx.x >> 5));
    const long rows = (long)wid * 16;
    if (rows >= N0) return;                 // scalar branch (wid in SGPR)
    const bool doR = (rows < N1);           // scalar condition (N1 % 16 == 0)
    const int  m   = lane & 15;
    const int  hi  = lane >> 4;

    // Hoist weight fragments (8 KB working set, L1-resident across waves).
    v16h bl[4], br[4];
#pragma unroll
    for (int t = 0; t < 4; ++t)
        for (int s = 0; s < 16; ++s)
            bl[t][s] = (_Float16)Wl[(32 * t + (s < 8 ? 8 * hi + s : 16 + 8 * hi + (s - 8))) * 16 + m];
    if (doR) {
#pragma unroll
        for (int t = 0; t < 4; ++t)
            for (int s = 0; s < 16; ++s)
                br[t][s] = (_Float16)Wr[(32 * t + (s < 8 ? 8 * hi + s : 16 + 8 * hi + (s - 8))) * 16 + m];
    }

    const float* xrow = x + (rows + m) * 128 + (hi ? 8 : 0);
    v8f cl = {};
    v8f cr = {};
#pragma unroll
    for (int t = 0; t < 4; ++t) {
        const float* ap = xrow + 32 * t;
        float4 f0 = *(const float4*)(ap);
        float4 f1 = *(const float4*)(ap + 4);
        float4 f2 = *(const float4*)(ap + 16);
        float4 f3 = *(const float4*)(ap + 20);
        v16h a;
        a[0]=(_Float16)f0.x;  a[1]=(_Float16)f0.y;  a[2]=(_Float16)f0.z;  a[3]=(_Float16)f0.w;
        a[4]=(_Float16)f1.x;  a[5]=(_Float16)f1.y;  a[6]=(_Float16)f1.z;  a[7]=(_Float16)f1.w;
        a[8]=(_Float16)f2.x;  a[9]=(_Float16)f2.y;  a[10]=(_Float16)f2.z; a[11]=(_Float16)f2.w;
        a[12]=(_Float16)f3.x; a[13]=(_Float16)f3.y; a[14]=(_Float16)f3.z; a[15]=(_Float16)f3.w;

        cl = __builtin_amdgcn_wmma_f32_16x16x32_f16(false, a, false, bl[t],
                                                    (short)0, cl, false, false);
        if (doR)
            cr = __builtin_amdgcn_wmma_f32_16x16x32_f16(false, a, false, br[t],
                                                        (short)0, cr, false, false);
    }

    // C/D layout: lane -> N = lane&15, vgpr r -> M = r + 8*(lane>>4).
    const long base = rows + hi * 8;
#pragma unroll
    for (int r = 0; r < 8; ++r) y[(base + r) * 16 + m] = cl[r];
    if (doR) {
#pragma unroll
        for (int r = 0; r < 8; ++r) xr[(base + r) * 16 + m] = cr[r];
    }
}

// ---------------------------------------------------------------------------
// Kernel 2 (shared by both layers): per-edge scatter-add of 16-wide rows plus
// float edge counts. 16 threads per edge; one cacheline gather per edge.
// Accumulators are L2-resident (<=12.8 MB vs 192 MB L2).
// ---------------------------------------------------------------------------
__global__ __launch_bounds__(256) void k_scatter(const float* __restrict__ vals,
                                                 const int* __restrict__ src,
                                                 const int* __restrict__ dst,
                                                 float* __restrict__ sums,
                                                 float* __restrict__ cnt,
                                                 int nedges, int ncomp) {
    long t = (long)blockIdx.x * blockDim.x + threadIdx.x;
    long e = t >> 4;
    int  c = (int)(t & 15);
    if (e >= (long)nedges) return;
    int s = src[e];
    int d = dst[e];
    if (c < ncomp) atomicAdd(&sums[(long)d * 16 + c], vals[(long)s * 16 + c]);
    if (c == 0)   atomicAdd(&cnt[d], 1.0f);
}

// ---------------------------------------------------------------------------
// Kernel 3: h = relu(sum/max(cnt,1) + b1 + xr)
// ---------------------------------------------------------------------------
__global__ __launch_bounds__(256) void k_node1(const float* __restrict__ sums,
                                               const float* __restrict__ cnt,
                                               const float* __restrict__ xr,
                                               const float* __restrict__ b1,
                                               float* __restrict__ hout) {
    long t = (long)blockIdx.x * blockDim.x + threadIdx.x;
    if (t >= (long)N1 * 16) return;
    long i = t >> 4;
    int  c = (int)(t & 15);
    float mcnt = fmaxf(cnt[i], 1.0f);
    float v = sums[t] / mcnt + b1[c] + xr[t];
    hout[t] = fmaxf(v, 0.0f);
}

// ---------------------------------------------------------------------------
// Kernel 4: z = h @ W2l for all N1 rows; r2 = h @ W2r for rows < N2.
// K=16 zero-padded to 32; N=14 zero-padded to 16. One WMMA per product.
// Same SGPR-uniform branching discipline as k_gemm1.
// ---------------------------------------------------------------------------
__global__ __launch_bounds__(256) void k_gemm2(const float* __restrict__ h,
                                               const float* __restrict__ Wl,
                                               const float* __restrict__ Wr,
                                               float* __restrict__ z,
                                               float* __restrict__ r2) {
    const int  lane = threadIdx.x & 31;
    const int  wid  = __builtin_amdgcn_readfirstlane(
                          blockIdx.x * (blockDim.x >> 5) + (threadIdx.x >> 5));
    const long rows = (long)wid * 16;
    if (rows >= N1) return;                 // scalar branch
    const bool doR = (rows < N2);           // scalar condition
    const int  m   = lane & 15;
    const int  hi  = lane >> 4;

    v16h bl, br;
#pragma unroll
    for (int n = 0; n < 16; ++n) { bl[n] = (_Float16)0.f; br[n] = (_Float16)0.f; }
    if (m < 14) {                           // column N=m; elements 8..15 (K 16..31) stay zero (padding);
        const float* wl = Wl + m;           // divergence OK: no WMMA inside
        const float* wr = Wr + m;
#pragma unroll
        for (int s = 0; s < 8; ++s) { bl[s] = (_Float16)wl[(8 * hi + s) * 14]; br[s] = (_Float16)wr[(8 * hi + s) * 14]; }
    }

    const float* hp = h + (rows + m) * 16 + hi * 8;
    float4 f0 = *(const float4*)(hp);
    float4 f1 = *(const float4*)(hp + 4);
    v16h a;
    a[0]=(_Float16)f0.x; a[1]=(_Float16)f0.y; a[2]=(_Float16)f0.z; a[3]=(_Float16)f0.w;
    a[4]=(_Float16)f1.x; a[5]=(_Float16)f1.y; a[6]=(_Float16)f1.z; a[7]=(_Float16)f1.w;
#pragma unroll
    for (int i = 8; i < 16; ++i) a[i] = (_Float16)0.f;   // K 16..31 padding

    v8f cl = {};
    v8f cr = {};
    cl = __builtin_amdgcn_wmma_f32_16x16x32_f16(false, a, false, bl, (short)0, cl, false, false);
    if (doR)
        cr = __builtin_amdgcn_wmma_f32_16x16x32_f16(false, a, false, br, (short)0, cr, false, false);

    const long base = rows + hi * 8;
#pragma unroll
    for (int r = 0; r < 8; ++r) z[(base + r) * 16 + m] = cl[r];
    if (doR) {
#pragma unroll
        for (int r = 0; r < 8; ++r) {
            long row = base + r;
            if (row < N2) r2[row * 16 + m] = cr[r];   // ragged 25000/16 boundary
        }
    }
}

// ---------------------------------------------------------------------------
// Kernel 5: out = log_softmax(sum/max(cnt,1) + b2 + r2) over 14 classes.
// ---------------------------------------------------------------------------
__global__ __launch_bounds__(256) void k_final(const float* __restrict__ sums,
                                               const float* __restrict__ cnt,
                                               const float* __restrict__ r2,
                                               const float* __restrict__ b2,
                                               float* __restrict__ out) {
    int i = blockIdx.x * blockDim.x + threadIdx.x;
    if (i >= N2) return;
    float mcnt = fmaxf(cnt[i], 1.0f);
    float v[14];
    float mx = -3.4e38f;
#pragma unroll
    for (int c = 0; c < 14; ++c) {
        v[c] = sums[(long)i * 16 + c] / mcnt + b2[c] + r2[(long)i * 16 + c];
        mx = fmaxf(mx, v[c]);
    }
    float s = 0.f;
#pragma unroll
    for (int c = 0; c < 14; ++c) s += __expf(v[c] - mx);
    float lse = __logf(s) + mx;
#pragma unroll
    for (int c = 0; c < 14; ++c) out[(long)i * 14 + c] = v[c] - lse;
}

// ---------------------------------------------------------------------------
extern "C" void kernel_launch(void* const* d_in, const int* in_sizes, int n_in,
                              void* d_out, int out_size, void* d_ws, size_t ws_size,
                              hipStream_t stream) {
    const float* x   = (const float*)d_in[0];
    const float* W1l = (const float*)d_in[1];
    const float* W1r = (const float*)d_in[2];
    const float* b1  = (const float*)d_in[3];
    const float* W2l = (const float*)d_in[4];
    const float* W2r = (const float*)d_in[5];
    const float* b2  = (const float*)d_in[6];
    const int*   src1 = (const int*)d_in[7];
    const int*   dst1 = (const int*)d_in[8];
    const int*   src2 = (const int*)d_in[9];
    const int*   dst2 = (const int*)d_in[10];
    float* out = (float*)d_out;

    // Workspace carve-out (~115 MB), 256B aligned chunks.
    char* p = (char*)d_ws;
    auto take = [&](size_t bytes) -> char* {
        char* q = p;
        p += (bytes + 255) & ~(size_t)255;
        return q;
    };
    float* y1   = (float*)take((size_t)N0 * 16 * 4);                      // x@W1l
    float* xr1  = (float*)take((size_t)N1 * 16 * 4);                      // x[:N1]@W1r
    float* sum1 = (float*)take((size_t)N1 * 16 * 4 + (size_t)N1 * 4);     // sums + counts
    float* cnt1 = sum1 + (size_t)N1 * 16;
    float* hbuf = (float*)take((size_t)N1 * 16 * 4);                      // layer-1 output
    float* z1   = (float*)take((size_t)N1 * 16 * 4);                      // h@W2l
    float* r2b  = (float*)take((size_t)N2 * 16 * 4);                      // h[:N2]@W2r
    float* sum2 = (float*)take((size_t)N2 * 16 * 4 + (size_t)N2 * 4);
    float* cnt2 = sum2 + (size_t)N2 * 16;

    // Zero the atomic accumulators every call (deterministic, capture-safe).
    hipMemsetAsync(sum1, 0, (size_t)N1 * 16 * 4 + (size_t)N1 * 4, stream);
    hipMemsetAsync(sum2, 0, (size_t)N2 * 16 * 4 + (size_t)N2 * 4, stream);

    // Layer 1
    k_gemm1<<<(N0 / 16 + 7) / 8, 256, 0, stream>>>(x, W1l, W1r, y1, xr1);
    k_scatter<<<(int)(((long)E1 * 16 + 255) / 256), 256, 0, stream>>>(
        y1, src1, dst1, sum1, cnt1, E1, 16);
    k_node1<<<((long)N1 * 16 + 255) / 256, 256, 0, stream>>>(sum1, cnt1, xr1, b1, hbuf);

    // Layer 2
    k_gemm2<<<(N1 / 16 + 7) / 8, 256, 0, stream>>>(hbuf, W2l, W2r, z1, r2b);
    k_scatter<<<(int)(((long)E2 * 16 + 255) / 256), 256, 0, stream>>>(
        z1, src2, dst2, sum2, cnt2, E2, 14);
    k_final<<<(N2 + 255) / 256, 256, 0, stream>>>(sum2, cnt2, r2b, b2, out);
}
